// SimpleModel_69578470195787
// MI455X (gfx1250) — matrix operand order repaired and checked by reading, not compiled, not run on hardware
//
#include <hip/hip_runtime.h>
#include <stdint.h>

// ---------------------------------------------------------------------------
// MI455X (gfx1250, wave32) implementation.
// All matmuls use v_wmma_f32_16x16x32_bf16 with fp32 accumulation.
// Fragments are loaded straight from global memory (no LDS tiling needed):
//   A (MxK row-major bf16): per-lane two contiguous 16B chunks.
//   B (NxK n-major  bf16): per-lane two contiguous 16B chunks.
// The patch-embed GEMM builds its A fragments directly from x (fp32) with
// the BN normalize fused in, using a channel-major K permutation so each
// 8-half WMMA chunk is a contiguous 8-float run of x. It is K-split 4x into
// per-split partial buffers (deterministic, no atomics) to fill the chip.
// sched_barrier(0) keeps all fragment loads hoisted ahead of the WMMA burst.
// ---------------------------------------------------------------------------

typedef __attribute__((ext_vector_type(16))) __bf16 v16bf;
typedef __attribute__((ext_vector_type(2)))  __bf16 v2bf;
typedef __attribute__((ext_vector_type(8)))  float  v8f;

#define BB   256
#define EE   128
#define PDp  12288          // C*PS*PS
#define HWp  65536          // H*W
#define NPCd 16777216.0     // per-channel element count B*H*W
#define KSPLIT 4
#define KSLICE 3072         // PDp / KSPLIT

union FragAB { v16bf v; uint4 u[2]; unsigned int w[8]; unsigned short h[16]; };

__device__ __forceinline__ void sched_fence() {
#if __has_builtin(__builtin_amdgcn_sched_barrier)
  __builtin_amdgcn_sched_barrier(0);
#endif
}

__device__ __forceinline__ unsigned short f2bf(float f) {
  unsigned int u = __float_as_uint(f);
  u += 0x7FFFu + ((u >> 16) & 1u);           // round to nearest even
  return (unsigned short)(u >> 16);
}

// pack two floats -> packed bf16x2 (lo in [15:0], hi in [31:16]).
// Hardware packer if present, else cheap round-half-up (4 VALU per pair).
__device__ __forceinline__ unsigned int pk2bf_fast(float lo, float hi) {
#if __has_builtin(__builtin_amdgcn_cvt_pk_bf16_f32)
  union { v2bf v; unsigned int u; } c;
  c.v = __builtin_amdgcn_cvt_pk_bf16_f32(lo, hi);
  return c.u;
#else
  unsigned int ul = __float_as_uint(lo) + 0x8000u;
  unsigned int uh = __float_as_uint(hi) + 0x8000u;
  return (ul >> 16) | (uh & 0xFFFF0000u);
#endif
}

// ---------------- BN statistics ----------------
__global__ void k_zero_stats(double* stats) {
  if (threadIdx.x < 8) stats[threadIdx.x] = 0.0;
}

__global__ void k_stats(const float* __restrict__ x, double* __restrict__ stats) {
  const int blk = blockIdx.x;                 // b*3 + c
  const int c = blk % 3;
  const float4* p = (const float4*)(x + (size_t)blk * HWp);
  float s = 0.f, q = 0.f;
  for (int i = 0; i < 64; ++i) {
    float4 v = p[threadIdx.x + i * 256];
    s += v.x + v.y + v.z + v.w;
    q += v.x * v.x + v.y * v.y + v.z * v.z + v.w * v.w;
  }
  __shared__ float ls[256], lq[256];
  ls[threadIdx.x] = s; lq[threadIdx.x] = q;
  __syncthreads();
  for (int off = 128; off > 0; off >>= 1) {
    if (threadIdx.x < off) {
      ls[threadIdx.x] += ls[threadIdx.x + off];
      lq[threadIdx.x] += lq[threadIdx.x + off];
    }
    __syncthreads();
  }
  if (threadIdx.x == 0) {
    atomicAdd(&stats[c], (double)ls[0]);
    atomicAdd(&stats[3 + c], (double)lq[0]);
  }
}

__global__ void k_finalize_stats(const double* __restrict__ stats,
                                 const float* __restrict__ bnw,
                                 const float* __restrict__ bnb,
                                 float* __restrict__ ss) {
  int c = threadIdx.x;
  if (c >= 3) return;
  double mean = stats[c] / NPCd;
  double var  = stats[3 + c] / NPCd - mean * mean;
  float inv   = (float)(1.0 / sqrt(var + 1e-5));
  float sc    = bnw[c] * inv;
  ss[c]     = sc;
  ss[3 + c] = bnb[c] - (float)mean * sc;
}

// ---------------- weight conversions (fp32 -> bf16, n-major) --------------
// pe_wbf[e][d'] with d' = c*4096 + py*64 + px  (channel-major K permutation)
__global__ void k_conv_pe(const float* __restrict__ pe_w,
                          unsigned short* __restrict__ pewbf) {
  int o = blockIdx.x * 256 + threadIdx.x;     // 128*12288
  int e  = o / PDp;
  int dp = o - e * PDp;
  int c   = dp >> 12;
  int pyx = dp & 4095;
  pewbf[o] = f2bf(pe_w[(size_t)e * PDp + pyx * 3 + c]);
}

// Wuv: 256 rows x 128 K. Row n'<128 -> corr_w[n'][0:128]; n'>=128 -> corr_w[n'-128][128:256]
__global__ void k_conv_corr(const float* __restrict__ cw,
                            unsigned short* __restrict__ out) {
  int o = blockIdx.x * 256 + threadIdx.x;     // 256*128
  int np = o >> 7, k = o & 127;
  int e = np & 127, half = np >> 7;
  out[o] = f2bf(cw[e * 256 + half * 128 + k]);
}

__global__ void k_conv_plain(const float* __restrict__ in,
                             unsigned short* __restrict__ out) {
  int o = blockIdx.x * 256 + threadIdx.x;     // 128*256
  out[o] = f2bf(in[o]);
}

// xc row b*16 + 0 = cls_token + pos_embed[0]
__global__ void k_xc_cls(const float* __restrict__ cls,
                         const float* __restrict__ pos,
                         unsigned short* __restrict__ xcbf) {
  int o = blockIdx.x * 256 + threadIdx.x;     // B*128
  int b = o >> 7, n = o & 127;
  xcbf[(size_t)(b * 16) * EE + n] = f2bf(cls[n] + pos[n]);
}

// ---------------- patch-embed GEMM (fused BN, K-split) --------------------
// M=4096 patches, N=128, K=12288 split into KSPLIT slices of KSLICE.
// Grid (32, KSPLIT), 256 threads = 8 waves. Wave w owns rows w*16..w*16+15
// of the block tile and ALL 8 N-subtiles: each x element packed once per WG.
// Each split writes a private fp32 partial buffer (deterministic reduction).
__global__ __launch_bounds__(256) void k_gemm_embed(
    const float* __restrict__ x, const float* __restrict__ ss,
    const unsigned short* __restrict__ pewbf, float* __restrict__ embp) {
  const int tid  = threadIdx.x;
  const int wave = tid >> 5, lane = tid & 31;
  const int hi   = lane >> 4, l15 = lane & 15;
  const int mrow0 = blockIdx.x * 128 + wave * 16;   // wave's 16 rows
  const int ks    = blockIdx.y;                     // K-split index
  const int kbeg  = ks * KSLICE, kend = kbeg + KSLICE;

  const float sc0 = ss[0], sc1 = ss[1], sc2 = ss[2];
  const float sh0 = ss[3], sh1 = ss[4], sh2 = ss[5];

  const int r  = mrow0 + l15;                 // patch row for this lane
  const int b_ = r >> 4, gh = (r >> 2) & 3, gw = r & 3;
  const int rbase = b_ * 196608 + gh * 16384 + gw * 64;  // b*3*HW + gh*64*256 + gw*64

  v8f acc[8];
  v8f zero = {};
  for (int t8 = 0; t8 < 8; ++t8) acc[t8] = zero;

  for (int kb = kbeg; kb < kend; kb += 32) {
    FragAB a;
    {
      int k0 = kb + hi * 8;
      for (int r2 = 0; r2 < 2; ++r2) {        // halves 0..7 then 8..15
        int k  = k0 + r2 * 16;                // contiguous 8-float run of x
        int c  = k >> 12, py = (k >> 6) & 63, px = k & 63;
        const float4* p = (const float4*)(x + (size_t)rbase + c * HWp + py * 256 + px);
        float4 f0 = p[0], f1 = p[1];
        float s = (c == 0) ? sc0 : (c == 1) ? sc1 : sc2;
        float t = (c == 0) ? sh0 : (c == 1) ? sh1 : sh2;
        int wb = r2 * 4;
        a.w[wb + 0] = pk2bf_fast(f0.x * s + t, f0.y * s + t);
        a.w[wb + 1] = pk2bf_fast(f0.z * s + t, f0.w * s + t);
        a.w[wb + 2] = pk2bf_fast(f1.x * s + t, f1.y * s + t);
        a.w[wb + 3] = pk2bf_fast(f1.z * s + t, f1.w * s + t);
      }
    }
    FragAB bm[8];
    for (int t8 = 0; t8 < 8; ++t8) {
      int n = t8 * 16 + l15;
      const uint4* p = (const uint4*)(pewbf + (size_t)n * PDp + kb + hi * 8);
      bm[t8].u[0] = p[0]; bm[t8].u[1] = p[2];
    }
    sched_fence();                            // keep loads hoisted & distinct
    for (int t8 = 0; t8 < 8; ++t8)
      acc[t8] = __builtin_amdgcn_wmma_f32_16x16x32_bf16(
          false, a.v, false, bm[t8].v, (short)0, acc[t8], false, false);
    sched_fence();
  }

  // store fp32 partial tile for this K-slice
  float* dst = embp + (size_t)ks * 4096 * EE;
  for (int t8 = 0; t8 < 8; ++t8) {
    int col = t8 * 16 + l15;
    for (int v = 0; v < 8; ++v) {
      int row = mrow0 + v + 8 * hi;
      dst[(size_t)row * EE + col] = acc[t8][v];
    }
  }
}

// Deterministic reduction of the KSPLIT partials + pe_b + pos_embed routing.
// token t = patch+1: t<16 -> xc (bf16); t==16 -> "last" output row (fp32).
__global__ void k_emb_fin(const float* __restrict__ embp,
                          const float* __restrict__ pe_b,
                          const float* __restrict__ pos,
                          unsigned short* __restrict__ xcbf,
                          float* __restrict__ out) {
  int o = blockIdx.x * 256 + threadIdx.x;     // 4096*128
  int row = o >> 7, col = o & 127;
  float val = embp[o];
  for (int ksp = 1; ksp < KSPLIT; ++ksp) val += embp[(size_t)ksp * 4096 * EE + o];
  int b = row >> 4, pch = row & 15, t = pch + 1;
  val += pe_b[col] + pos[t * EE + col];
  if (pch < 15)
    xcbf[(size_t)(b * 16 + t) * EE + col] = f2bf(val);
  else
    out[((size_t)b * 17 + 16) * EE + col] = val;
}

// ---------------- generic bf16 GEMM  D = A(MxK) * B(NxK)^T (+bias) --------
// mode 0: out fp32, leading dim ldOut, no remap (UV; bias==nullptr)
// mode 1: out bf16, ld 128 (reduction steps 1-3)
// mode 2: out fp32 into d_out with row remap m -> (m/16)*17 + m%16 (final step)
__global__ __launch_bounds__(256) void k_gemm_bf(
    const unsigned short* __restrict__ A, const unsigned short* __restrict__ Bm,
    const float* __restrict__ bias, float* __restrict__ outf,
    unsigned short* __restrict__ outbf, int K, int ldOut, int mode) {
  const int tid  = threadIdx.x;
  const int wave = tid >> 5, lane = tid & 31;
  const int hi   = lane >> 4, l15 = lane & 15;
  const int m0   = blockIdx.x * 128 + (wave & 3) * 32;
  const int nB   = blockIdx.y * 128;
  const int n0   = nB + (wave >> 2) * 64;

  v8f acc[2][4];
  v8f zero = {};
  for (int s2 = 0; s2 < 2; ++s2)
    for (int t4 = 0; t4 < 4; ++t4) acc[s2][t4] = zero;

  for (int kb = 0; kb < K; kb += 32) {
    FragAB a[2];
    for (int s2 = 0; s2 < 2; ++s2) {
      const uint4* p = (const uint4*)(A + (size_t)(m0 + s2 * 16 + l15) * K + kb + hi * 8);
      a[s2].u[0] = p[0]; a[s2].u[1] = p[2];   // +16 halves
    }
    FragAB bm[4];
    for (int t4 = 0; t4 < 4; ++t4) {
      const uint4* p = (const uint4*)(Bm + (size_t)(n0 + t4 * 16 + l15) * K + kb + hi * 8);
      bm[t4].u[0] = p[0]; bm[t4].u[1] = p[2];
    }
    sched_fence();                            // keep loads hoisted & distinct
    for (int s2 = 0; s2 < 2; ++s2)
      for (int t4 = 0; t4 < 4; ++t4)
        acc[s2][t4] = __builtin_amdgcn_wmma_f32_16x16x32_bf16(
            false, a[s2].v, false, bm[t4].v, (short)0, acc[s2][t4], false, false);
    sched_fence();
  }

  for (int t4 = 0; t4 < 4; ++t4) {
    int col = n0 + t4 * 16 + l15;
    float bv = bias ? bias[col & 127] : 0.f;
    for (int s2 = 0; s2 < 2; ++s2) {
      for (int v = 0; v < 8; ++v) {
        int row = m0 + s2 * 16 + v + 8 * hi;
        float val = acc[s2][t4][v] + bv;
        if (mode == 0)
          outf[(size_t)row * ldOut + col] = val;
        else if (mode == 1)
          outbf[(size_t)row * EE + col] = f2bf(val);
        else
          outf[(size_t)((row >> 4) * 17 + (row & 15)) * EE + col] = val;
      }
    }
  }
}

// h0[b,i,j,n] = U[b*16+i][n] + V[b*16+j][n] + corr_b[n]   (bf16, paired stores)
__global__ void k_h0(const float* __restrict__ UV, const float* __restrict__ corr_b,
                     unsigned int* __restrict__ h0) {
  int o = blockIdx.x * 256 + threadIdx.x;     // 256*16*16*64 pair-units
  int n2 = o & 63;                            // pair index; n = 2*n2
  int j = (o >> 6) & 15;
  int i = (o >> 10) & 15;
  int b = o >> 14;
  int n = n2 * 2;
  const float* Urow = UV + (size_t)(b * 16 + i) * 256;
  const float* Vrow = UV + (size_t)(b * 16 + j) * 256 + 128;
  float lo = Urow[n]     + Vrow[n]     + corr_b[n];
  float hi = Urow[n + 1] + Vrow[n + 1] + corr_b[n + 1];
  h0[o] = ((unsigned int)f2bf(lo)) | (((unsigned int)f2bf(hi)) << 16);
}

// ---------------- launcher ----------------
extern "C" void kernel_launch(void* const* d_in, const int* in_sizes, int n_in,
                              void* d_out, int out_size, void* d_ws, size_t ws_size,
                              hipStream_t stream) {
  (void)in_sizes; (void)n_in; (void)out_size; (void)ws_size;
  const float* x       = (const float*)d_in[0];
  const float* bnw     = (const float*)d_in[1];
  const float* bnb     = (const float*)d_in[2];
  const float* pe_w    = (const float*)d_in[3];
  const float* pe_b    = (const float*)d_in[4];
  const float* pos     = (const float*)d_in[5];
  const float* cls     = (const float*)d_in[6];
  const float* corr_w  = (const float*)d_in[7];
  const float* corr_b  = (const float*)d_in[8];
  const float* cross_w = (const float*)d_in[9];
  const float* cross_b = (const float*)d_in[10];
  float* out = (float*)d_out;
  char*  ws  = (char*)d_ws;

  // workspace layout (~40.2 MB total)
  double*         stats  = (double*)(ws + 0);
  float*          ssf    = (float*)(ws + 64);
  unsigned short* pewbf  = (unsigned short*)(ws + 256);        // 3,145,728 B
  unsigned short* wuv    = (unsigned short*)(ws + 3145984);    //    65,536 B
  unsigned short* wcross = (unsigned short*)(ws + 3211520);    //    65,536 B
  unsigned short* xcbf   = (unsigned short*)(ws + 3277056);    // 1,048,576 B
  float*          UV     = (float*)(ws + 4325632);             // 4,194,304 B
  unsigned short* hA     = (unsigned short*)(ws + 8519936);    // 16,777,216 B
  unsigned short* hB     = (unsigned short*)(ws + 25297152);   //  8,388,608 B
  float*          embp   = (float*)(ws + 33685760);            //  8,388,608 B (4 partials)

  k_zero_stats<<<1, 32, 0, stream>>>(stats);
  k_stats<<<768, 256, 0, stream>>>(x, stats);
  k_finalize_stats<<<1, 4, 0, stream>>>(stats, bnw, bnb, ssf);
  k_conv_pe<<<6144, 256, 0, stream>>>(pe_w, pewbf);
  k_conv_corr<<<128, 256, 0, stream>>>(corr_w, wuv);
  k_conv_plain<<<128, 256, 0, stream>>>(cross_w, wcross);
  k_xc_cls<<<128, 256, 0, stream>>>(cls, pos, xcbf);
  // patch embed: M=4096 N=128 K=12288 (4-way K-split partials)
  k_gemm_embed<<<dim3(32, KSPLIT), 256, 0, stream>>>(x, ssf, pewbf, embp);
  // deterministic partial reduction + bias/pos routing (xc bf16 + last token)
  k_emb_fin<<<2048, 256, 0, stream>>>(embp, pe_b, pos, xcbf, out);
  // U|V = xc @ Wuv^T : M=4096 N=256 K=128
  k_gemm_bf<<<dim3(32, 2), 256, 0, stream>>>(xcbf, wuv, nullptr, UV, nullptr, 128, 256, 0);
  // h0 broadcast add (pair-packed bf16 stores): 4,194,304 pair units
  k_h0<<<16384, 256, 0, stream>>>(UV, corr_b, (unsigned int*)hA);
  // 4 reduction steps: (M_s x 256) @ cross_w^T, M_s: 32768 -> 4096
  k_gemm_bf<<<dim3(256, 1), 256, 0, stream>>>(hA, wcross, cross_b, nullptr, hB, 256, 128, 1);
  k_gemm_bf<<<dim3(128, 1), 256, 0, stream>>>(hB, wcross, cross_b, nullptr, hA, 256, 128, 1);
  k_gemm_bf<<<dim3(64, 1), 256, 0, stream>>>(hA, wcross, cross_b, nullptr, hB, 256, 128, 1);
  k_gemm_bf<<<dim3(32, 1), 256, 0, stream>>>(hB, wcross, cross_b, out, nullptr, 256, 128, 2);
}
